// CharLSTM_30399778521197
// MI455X (gfx1250) — compile-verified, matrix operand order checked by reading, not run on hardware
//
#include <hip/hip_runtime.h>
#include <math.h>

#define B_   64
#define T_   512
#define V_   100
#define E_   100
#define EPAD 128
#define H_   256
#define FH   1024
#define L_   8

typedef __attribute__((ext_vector_type(16))) _Float16 v16h;
typedef __attribute__((ext_vector_type(8)))  _Float16 v8h;
typedef __attribute__((ext_vector_type(8)))  float    v8f;

__device__ __forceinline__ float sigmoidf_(float x) {
    return 1.0f / (1.0f + __expf(-x));
}

__device__ __forceinline__ v8f wmma16(v16h a, v16h b, v8f c) {
    return __builtin_amdgcn_wmma_f32_16x16x32_f16(false, a, false, b,
                                                  (short)0, c, false, false);
}

// Load a 16x32 f16 fragment from row-major memory (A rows, or B columns = W rows).
// CDNA5 layout: lanes 0-15 hold row=lane, K={0..7,16..23}; lanes 16-31 row=lane-16,
// K={8..15,24..31}  ->  two 16-byte loads per lane.
__device__ __forceinline__ v16h load_frag(const _Float16* __restrict__ p, int ld) {
    const int lane = threadIdx.x & 31;
    const int row  = lane & 15;
    const int ko   = (lane >> 4) << 3;   // 0 or 8
    const _Float16* q = p + (size_t)row * ld + ko;
    v8h lo = *(const v8h*)(q);        // K = ko .. ko+7
    v8h hi = *(const v8h*)(q + 16);   // K = ko+16 .. ko+23
    v16h r;
#pragma unroll
    for (int i = 0; i < 8; ++i) { r[i] = lo[i]; r[i + 8] = hi[i]; }
    return r;
}

// Async global->LDS copy of 16 bytes (per lane), tracked by ASYNCcnt.
__device__ __forceinline__ void async_copy16(unsigned lds_off, const void* gaddr) {
    asm volatile("global_load_async_to_lds_b128 %0, %1, off"
                 :: "v"(lds_off), "v"(gaddr)
                 : "memory");
}
__device__ __forceinline__ void wait_async0() {
    asm volatile("s_wait_asynccnt 0x0" ::: "memory");
}

// ---------------- small prep kernels ----------------

__global__ void k_f32_to_f16(const float* __restrict__ s, _Float16* __restrict__ d, int n) {
    int i = blockIdx.x * blockDim.x + threadIdx.x;
    if (i < n) d[i] = (_Float16)s[i];
}

__global__ void k_pad_wih0(const float* __restrict__ s, _Float16* __restrict__ d) {
    int i = blockIdx.x * blockDim.x + threadIdx.x;   // FH*EPAD
    if (i >= FH * EPAD) return;
    int r = i >> 7, c = i & 127;
    d[i] = (_Float16)(c < E_ ? s[r * E_ + c] : 0.0f);
}

__global__ void k_pad_fcw(const float* __restrict__ s, _Float16* __restrict__ d) {
    int i = blockIdx.x * blockDim.x + threadIdx.x;   // 112*H_
    if (i >= 112 * H_) return;
    int r = i >> 8, c = i & 255;
    d[i] = (_Float16)(r < V_ ? s[r * H_ + c] : 0.0f);
}

__global__ void k_bsum(const float* __restrict__ bih, const float* __restrict__ bhh,
                       float* __restrict__ bsum, int n) {
    int i = blockIdx.x * blockDim.x + threadIdx.x;
    if (i < n) bsum[i] = bih[i] + bhh[i];
}

__global__ void k_zero_u32(unsigned* __restrict__ p, int n) {
    int i = blockIdx.x * blockDim.x + threadIdx.x;
    if (i < n) p[i] = 0u;
}

__global__ void k_embed(const int* __restrict__ x, const float* __restrict__ embed,
                        _Float16* __restrict__ act) {
    int bt = blockIdx.x;           // 0 .. B*T-1
    int e  = threadIdx.x;          // 0 .. 127
    int tok = x[bt];
    act[(size_t)bt * EPAD + e] = (_Float16)(e < E_ ? embed[(size_t)tok * E_ + e] : 0.0f);
}

// ---------------- layer-pipelined persistent LSTM ----------------
// Grid = 8 blocks (one layer each, each on its own WGP), 1024 threads = 32 waves.
// Wave w: j-tile jt=w&15 (16 hidden cols), batch-tile pair mtBase=(w>>4)*2.
// Each wave computes gate N-tiles {jt, jt+16, jt+32, jt+48} so i/f/g/o for a
// column land in the same lane registers. Per step:
//   g = x_t @ W_ih^T + h_{t-1} @ W_hh^T + (b_ih + b_hh)
// x_t slab (64 x Kx f16) is staged global->LDS with async-to-LDS loads.

__global__ void __launch_bounds__(1024)
k_lstm_pipe(const _Float16* __restrict__ embBuf,   // [B*T][EPAD]
            _Float16* __restrict__ acts,           // [L][B][T][H]
            const _Float16* __restrict__ wih0,     // [FH][EPAD]
            const _Float16* __restrict__ wih,      // [L-1][FH][H]
            const _Float16* __restrict__ whh,      // [L][FH][H]
            const float* __restrict__ bsum,        // [L][FH]
            unsigned* __restrict__ prog,           // [L] progress flags
            float* __restrict__ hfin,              // [L][B][H]
            float* __restrict__ cfin) {            // [L][B][H]
    __shared__ _Float16 hbuf[B_ * H_];      // 32 KB: h_{t-1} of this layer
    __shared__ _Float16 xbuf[B_ * H_];      // 32 KB: x_t slab from layer below

    const int l       = blockIdx.x;
    const int kxShift = (l == 0) ? 7 : 8;
    const int Kx      = 1 << kxShift;
    const int kxSteps = Kx >> 5;            // 4 or 8
    const _Float16* xsrc = (l == 0) ? embBuf : acts + (size_t)(l - 1) * B_ * T_ * H_;
    const _Float16* wx   = (l == 0) ? wih0  : wih  + (size_t)(l - 1) * FH * H_;
    const _Float16* wh   = whh + (size_t)l * FH * H_;
    const float*    bs   = bsum + (size_t)l * FH;
    _Float16*       aout = acts + (size_t)l * B_ * T_ * H_;

    const int tid = threadIdx.x;
    for (int i = tid; i < B_ * H_; i += 1024) hbuf[i] = (_Float16)0.0f;
    __syncthreads();

    const int w      = tid >> 5;
    const int lane   = tid & 31;
    const int jt     = w & 15;
    const int mtBase = (w >> 4) * 2;
    const int nlane  = lane & 15;
    const int m0     = (lane >> 4) << 3;
    const int j      = jt * 16 + nlane;

    float bvv[4];
#pragma unroll
    for (int g = 0; g < 4; ++g) bvv[g] = bs[g * 256 + j];

    float creg[2][8];
    float hlast[2][8];
#pragma unroll
    for (int mi = 0; mi < 2; ++mi)
#pragma unroll
        for (int r = 0; r < 8; ++r) { creg[mi][r] = 0.0f; hlast[mi][r] = 0.0f; }

    const unsigned xbase = (unsigned)(uintptr_t)(&xbuf[0]);
    const int halvesTotal = B_ * Kx;        // 8192 or 16384

    for (int t = 0; t < T_; ++t) {
        // ---- wait for the producer layer to publish x_t ----
        if (l > 0) {
            if (tid == 0) {
                while (__hip_atomic_load(prog + (l - 1), __ATOMIC_ACQUIRE,
                                         __HIP_MEMORY_SCOPE_AGENT) < (unsigned)(t + 1)) {
                    __builtin_amdgcn_s_sleep(2);
                }
            }
            __syncthreads();                      // B1: flag seen, L0 invalidated
        }

        // ---- stage x_t slab into LDS with async-to-LDS loads ----
        for (int i = tid * 8; i < halvesTotal; i += 1024 * 8) {
            int row = i >> kxShift;
            int col = i - (row << kxShift);
            const _Float16* g = xsrc + (((size_t)row * T_ + t) << kxShift) + col;
            async_copy16(xbase + (unsigned)i * 2u, g);
        }

        // ---- init accumulators with folded biases ----
        v8f acc[2][4];
#pragma unroll
        for (int mi = 0; mi < 2; ++mi)
#pragma unroll
            for (int g = 0; g < 4; ++g)
#pragma unroll
                for (int r = 0; r < 8; ++r) acc[mi][g][r] = bvv[g];

        wait_async0();
        __syncthreads();                          // B2: xbuf ready, hbuf stable

        // ---- x_t @ W_ih^T ----
        for (int k = 0; k < kxSteps; ++k) {
            v16h a0 = load_frag(xbuf + ((16 * mtBase) << kxShift) + 32 * k, Kx);
            v16h a1 = load_frag(xbuf + ((16 * (mtBase + 1)) << kxShift) + 32 * k, Kx);
#pragma unroll
            for (int g = 0; g < 4; ++g) {
                v16h bb = load_frag(wx + (((size_t)(g * 256 + jt * 16)) << kxShift) + 32 * k, Kx);
                acc[0][g] = wmma16(a0, bb, acc[0][g]);
                acc[1][g] = wmma16(a1, bb, acc[1][g]);
            }
        }

        // ---- h_{t-1} @ W_hh^T ----
#pragma unroll
        for (int k = 0; k < 8; ++k) {
            v16h a0 = load_frag(hbuf + (16 * mtBase) * H_ + 32 * k, H_);
            v16h a1 = load_frag(hbuf + (16 * (mtBase + 1)) * H_ + 32 * k, H_);
#pragma unroll
            for (int g = 0; g < 4; ++g) {
                v16h bb = load_frag(wh + (size_t)(g * 256 + jt * 16) * H_ + 32 * k, H_);
                acc[0][g] = wmma16(a0, bb, acc[0][g]);
                acc[1][g] = wmma16(a1, bb, acc[1][g]);
            }
        }

        __syncthreads();                          // B3: done reading hbuf/xbuf

        // ---- gates, state update, publish h_t ----
#pragma unroll
        for (int mi = 0; mi < 2; ++mi) {
            const int b0 = 16 * (mtBase + mi) + m0;
#pragma unroll
            for (int r = 0; r < 8; ++r) {
                float ig = sigmoidf_(acc[mi][0][r]);
                float fg = sigmoidf_(acc[mi][1][r]);
                float gg = tanhf(acc[mi][2][r]);
                float og = sigmoidf_(acc[mi][3][r]);
                float cn = fg * creg[mi][r] + ig * gg;
                creg[mi][r] = cn;
                float hv = og * tanhf(cn);
                hlast[mi][r] = hv;
                int bi = b0 + r;
                _Float16 hh = (_Float16)hv;
                hbuf[bi * H_ + j] = hh;
                aout[((size_t)bi * T_ + t) * H_ + j] = hh;
            }
        }

        __threadfence();                          // my stores visible at agent scope
        __syncthreads();                          // B4: everyone fenced, hbuf = h_t
        if (tid == 0)
            __hip_atomic_store(prog + l, (unsigned)(t + 1), __ATOMIC_RELEASE,
                               __HIP_MEMORY_SCOPE_AGENT);
    }

    // ---- final h/c for this layer ----
#pragma unroll
    for (int mi = 0; mi < 2; ++mi)
#pragma unroll
        for (int r = 0; r < 8; ++r) {
            int bi = 16 * (mtBase + mi) + m0 + r;
            hfin[(size_t)bi * H_ + j] = hlast[mi][r];
            cfin[(size_t)bi * H_ + j] = creg[mi][r];
        }
}

// ---------------- final FC: logits = act * fc_w^T + fc_b ----------------

__global__ void k_fc(const _Float16* __restrict__ A,   // [B*T][H]
                     const _Float16* __restrict__ W,   // [112][H]
                     const float* __restrict__ bias,   // [V]
                     float* __restrict__ out) {        // [B*T][V]
    int wave = (blockIdx.x * blockDim.x + threadIdx.x) >> 5;
    int mt = wave / 7;     // 0..2047
    int nt = wave % 7;     // 0..6
    v8f acc = {};
#pragma unroll
    for (int k = 0; k < 8; ++k) {
        v16h a = load_frag(A + (size_t)(mt * 16) * H_ + 32 * k, H_);
        v16h b = load_frag(W + (size_t)(nt * 16) * H_ + 32 * k, H_);
        acc = wmma16(a, b, acc);
    }
    const int lane = threadIdx.x & 31;
    const int n  = nt * 16 + (lane & 15);
    const int m0 = mt * 16 + ((lane >> 4) << 3);
    if (n < V_) {
        float bn = bias[n];
#pragma unroll
        for (int r = 0; r < 8; ++r)
            out[(size_t)(m0 + r) * V_ + n] = acc[r] + bn;
    }
}

// ---------------- driver ----------------

extern "C" void kernel_launch(void* const* d_in, const int* in_sizes, int n_in,
                              void* d_out, int out_size, void* d_ws, size_t ws_size,
                              hipStream_t stream) {
    const int*   x     = (const int*)d_in[0];
    const float* embed = (const float*)d_in[1];
    const float* w_ih0 = (const float*)d_in[2];
    const float* w_ih  = (const float*)d_in[3];
    const float* w_hh  = (const float*)d_in[4];
    const float* b_ih  = (const float*)d_in[5];
    const float* b_hh  = (const float*)d_in[6];
    const float* fc_w  = (const float*)d_in[7];
    const float* fc_b  = (const float*)d_in[8];

    _Float16* embBuf = (_Float16*)d_ws;                          // B*T*EPAD    (8 MB)
    _Float16* acts   = embBuf + (size_t)B_ * T_ * EPAD;          // L*B*T*H     (128 MB)
    _Float16* wih0h  = acts + (size_t)L_ * B_ * T_ * H_;         // FH*EPAD
    _Float16* wihh   = wih0h + (size_t)FH * EPAD;                // 7*FH*H
    _Float16* whhh   = wihh + (size_t)7 * FH * H_;               // 8*FH*H
    _Float16* fcwh   = whhh + (size_t)8 * FH * H_;               // 112*H
    float*    bsum   = (float*)(fcwh + (size_t)112 * H_);        // L*FH
    unsigned* prog   = (unsigned*)(bsum + (size_t)L_ * FH);      // L flags

    float* logits = (float*)d_out;
    float* hfin   = logits + (size_t)B_ * T_ * V_;
    float* cfin   = hfin + (size_t)L_ * B_ * H_;

    // prep: weight conversion/padding, bias fold, flag reset, embedding gather
    k_pad_wih0<<<(FH * EPAD + 255) / 256, 256, 0, stream>>>(w_ih0, wih0h);
    k_f32_to_f16<<<(7 * FH * H_ + 255) / 256, 256, 0, stream>>>(w_ih, wihh, 7 * FH * H_);
    k_f32_to_f16<<<(8 * FH * H_ + 255) / 256, 256, 0, stream>>>(w_hh, whhh, 8 * FH * H_);
    k_pad_fcw<<<(112 * H_ + 255) / 256, 256, 0, stream>>>(fc_w, fcwh);
    k_bsum<<<(L_ * FH + 255) / 256, 256, 0, stream>>>(b_ih, b_hh, bsum, L_ * FH);
    k_zero_u32<<<1, 32, 0, stream>>>(prog, L_);
    k_embed<<<B_ * T_, EPAD, 0, stream>>>(x, embed, embBuf);

    // layer-pipelined persistent LSTM: 8 blocks, one per layer
    k_lstm_pipe<<<L_, 1024, 0, stream>>>(embBuf, acts, wih0h, wihh, whhh,
                                         bsum, prog, hfin, cfin);

    // final FC over layer-7 activations: 2048*7 tiles, 1 wave/tile
    k_fc<<<1792, 256, 0, stream>>>(acts + (size_t)7 * B_ * T_ * H_, fcwh, fc_b, logits);
}
